// GAT_41729902248227
// MI455X (gfx1250) — matrix operand order repaired and checked by reading, not compiled, not run on hardware
//
#include <hip/hip_runtime.h>

typedef __bf16 bf16;
typedef __attribute__((ext_vector_type(16))) bf16  v16bf;
typedef __attribute__((ext_vector_type(8)))  bf16  v8bf;
typedef __attribute__((ext_vector_type(8)))  float v8f;
typedef unsigned int u32;
typedef __attribute__((ext_vector_type(4))) u32 u32x4;
typedef __attribute__((ext_vector_type(4))) int i32x4;
typedef __attribute__((ext_vector_type(8))) int i32x8;

#define NN   50000   // nodes
#define EE   800000  // edges

#if __has_builtin(__builtin_amdgcn_tensor_load_to_lds)
#define USE_TDM 1
#else
#define USE_TDM 0
#endif

// scheduling-group hints: 0x008 = MFMA/WMMA, 0x100 = DS read
#if __has_builtin(__builtin_amdgcn_sched_group_barrier)
#define SGB(mask, n) __builtin_amdgcn_sched_group_barrier(mask, n, 0)
#else
#define SGB(mask, n)
#endif

// ---------------------------------------------------------------------------
// fp32 -> bf16 elementwise convert
// ---------------------------------------------------------------------------
__global__ void k_f2bf(const float* __restrict__ in, bf16* __restrict__ out, int count) {
    int i = blockIdx.x * blockDim.x + threadIdx.x;
    if (i < count) out[i] = (bf16)in[i];
}

// W [K x NO] fp32 (row-major) -> Wt [NO x K] bf16 (row-major)
__global__ void k_wtrans_bf16(const float* __restrict__ W, bf16* __restrict__ Wt,
                              int K, int NO) {
    int i = blockIdx.x * blockDim.x + threadIdx.x;
    if (i >= K * NO) return;
    int n = i % NO, k = i / NO;
    Wt[(size_t)n * K + k] = (bf16)W[i];
}

// ---------------------------------------------------------------------------
// WMMA GEMM: C[M x N] = A[M x K](bf16) * Bt[N x K](bf16)^T, fp32 accumulate.
// One wave computes a 16-row stripe across the full N width (NTILES tiles).
// K must be a multiple of 128 (processed in 128-wide chunks), N == 16*NTILES.
//
// Per chunk: wave 0 DMAs the whole B panel [N x 128] into LDS via the Tensor
// Data Mover (padded rows: 128 elems + 8 pad -> 272B stride, bank-conflict
// free ds_load_b128).  Each wave preloads its 4 A fragments in one clause.
// Per K-step, sched_group_barrier forces a software pipeline:
//   DS(2), { WMMA(1), DS(2) } x (NTILES-1), WMMA(1)
// so the next tile's B loads are in flight while the current WMMA executes.
//
// A frag per ISA 16-bit 16x32 layout: lanes 0-15 hold M=lane,
//   elems 0-7 = K(+0..7), elems 8-15 = K(+16..23); lanes 16-31 shift K by +8.
// B frag (32x16): lanes 0-15 hold N=lane, elems 0-7 = K(+0..7), elems 8-15 =
//   K(+16..23); lanes 16-31 shift K by +8.
// ---------------------------------------------------------------------------
#define KC    128          // K-chunk staged in LDS
#define LDSW  (KC + 8)     // padded LDS row stride in elements (272 B)

template <int NTILES>
__global__ __launch_bounds__(256) void k_gemm_bf16_wmma(
    const bf16* __restrict__ A, const bf16* __restrict__ Bt,
    float* __restrict__ C, int M, int K, int N)
{
    __shared__ bf16 bsm[NTILES * 16 * LDSW];

    const int lane = threadIdx.x & 31;
    const int wave = blockIdx.x * (blockDim.x >> 5) + (threadIdx.x >> 5);
    const int m0   = wave << 4;
    if (m0 >= M) return;                 // wave-uniform; barriers still complete

    int rowA = m0 + (lane & 15);
    if (rowA > M - 1) rowA = M - 1;      // clamp (stores are guarded)
    const int hi    = lane >> 4;         // 0 or 1
    const int koffA = hi * 8;
    const int koffB = hi * 8;
    const int nrow  = lane & 15;

    v8f acc[NTILES];
    const v8f zero = {0.f,0.f,0.f,0.f,0.f,0.f,0.f,0.f};
#pragma unroll
    for (int t = 0; t < NTILES; ++t) acc[t] = zero;

    const bf16* arow = A + (size_t)rowA * K;

    for (int kc = 0; kc < K; kc += KC) {
        __syncthreads();                 // previous chunk fully consumed
#if USE_TDM
        if ((threadIdx.x >> 5) == 0) {
            // ---- Tensor DMA descriptor (ISA 8.3/8.4), 2D tile, 2B elems ----
            unsigned long long ga = (unsigned long long)(uintptr_t)(Bt + kc);
            u32 ldsOff = (u32)(uintptr_t)(&bsm[0]);
            u32x4 g0;
            g0[0] = 1u;                                   // count=1, user mode
            g0[1] = ldsOff;                               // lds_addr
            g0[2] = (u32)(ga & 0xFFFFFFFFu);              // global_addr[31:0]
            g0[3] = (u32)((ga >> 32) & 0x1FFFFFFu) | (2u << 30); // [56:32]|type=2
            i32x8 g1;
            g1[0] = (1 << 16) | (1 << 20) | (5 << 22) | (3 << 25);
                    // data_size=2B | pad_enable | pad_interval=64dw | pad_amount=4dw
            g1[1] = (K & 0xFFFF) << 16;                   // tensor_dim0 lo
            g1[2] = ((K >> 16) & 0xFFFF) | ((N & 0xFFFF) << 16); // td0 hi | td1 lo
            g1[3] = ((N >> 16) & 0xFFFF) | (KC << 16);    // td1 hi | tile_dim0
            g1[4] = (N & 0xFFFF);                         // tile_dim1=N, tile_dim2=0
            g1[5] = K;                                    // tensor_dim0_stride lo
            g1[6] = 0;                                    // stride hi, dim1_stride lo
            g1[7] = 0;
            i32x4 gz = {0, 0, 0, 0};
#if defined(__clang_major__) && (__clang_major__ >= 23)
            i32x8 gz8 = {0, 0, 0, 0, 0, 0, 0, 0};
            __builtin_amdgcn_tensor_load_to_lds(g0, g1, gz, gz, gz8, 0);
#else
            __builtin_amdgcn_tensor_load_to_lds(g0, g1, gz, gz, 0);
#endif
            __builtin_amdgcn_s_wait_tensorcnt(0);
        }
#else
        // cooperative fallback: stage B chunk with vector copies
        for (int idx = threadIdx.x; idx < NTILES * 16 * (KC / 8); idx += (int)blockDim.x) {
            int r  = idx / (KC / 8);
            int c8 = idx % (KC / 8);
            *(v8bf*)&bsm[r * LDSW + c8 * 8] =
                *(const v8bf*)(Bt + (size_t)r * K + kc + c8 * 8);
        }
#endif
        __syncthreads();                 // B panel visible to all waves

        // preload the chunk's 4 A fragments in one clause (single load wait)
        union AV { v16bf v; v8bf h[2]; } a[4];
#pragma unroll
        for (int ks = 0; ks < 4; ++ks) {
            a[ks].h[0] = *(const v8bf*)(arow + kc + ks * 32 + koffA);
            a[ks].h[1] = *(const v8bf*)(arow + kc + ks * 32 + koffA + 16);
        }
#pragma unroll
        for (int ks = 0; ks < 4; ++ks) {
            union BV { v16bf v; v8bf h[2]; } b[NTILES];
#pragma unroll
            for (int t = 0; t < NTILES; ++t) {
                const int ro = (t * 16 + nrow) * LDSW + ks * 32 + koffB;
                b[t].h[0] = *(const v8bf*)&bsm[ro];
                b[t].h[1] = *(const v8bf*)&bsm[ro + 16];
            }
#pragma unroll
            for (int t = 0; t < NTILES; ++t)
                acc[t] = __builtin_amdgcn_wmma_f32_16x16x32_bf16(
                    false, a[ks].v, false, b[t].v, (short)0, acc[t], false, false);

            // force the software pipeline: next tile's DS reads issue before
            // the current WMMA so s_wait_dscnt covers already-in-flight loads
            SGB(0x100, 2);
#pragma unroll
            for (int t = 0; t < NTILES - 1; ++t) {
                SGB(0x008, 1);
                SGB(0x100, 2);
            }
            SGB(0x008, 1);
        }
    }

    // C/D layout: VGPR j, lanes 0-15 -> M=j, lanes 16-31 -> M=j+8, N=lane&15
    const int mbase = m0 + hi * 8;
#pragma unroll
    for (int t = 0; t < NTILES; ++t) {
#pragma unroll
        for (int j = 0; j < 8; ++j) {
            int m = mbase + j;
            if (m < M) C[(size_t)m * N + (t << 4) + nrow] = acc[t][j];
        }
    }
}

// ---------------------------------------------------------------------------
// Per-(node, head): attention logits el/er, init emax/denom/agg
// ---------------------------------------------------------------------------
__global__ void k_node_prep(const float* __restrict__ f,
                            const float* __restrict__ al,
                            const float* __restrict__ ar,
                            float* __restrict__ el, float* __restrict__ er,
                            float* __restrict__ emax, float* __restrict__ denom,
                            float* __restrict__ agg, int N, int H, int D)
{
    int i = blockIdx.x * blockDim.x + threadIdx.x;
    if (i >= N * H) return;
    int h = i % H, n = i / H;
    const float* fr  = f  + (size_t)n * H * D + (size_t)h * D;
    const float* alr = al + (size_t)h * D;
    const float* arr = ar + (size_t)h * D;
    float ev = 0.f, rv = 0.f;
    for (int d = 0; d < D; ++d) { float fv = fr[d]; ev += fv * alr[d]; rv += fv * arr[d]; }
    el[i] = ev; er[i] = rv;
    emax[i]  = -3.0e38f;
    denom[i] = 0.f;
    float* ag = agg + (size_t)n * H * D + (size_t)h * D;
    for (int d = 0; d < D; ++d) ag[d] = 0.f;
}

__device__ inline void atomicMaxF(float* addr, float val) {
    int* ia  = (int*)addr;
    int  cur = __float_as_int(*addr);
    while (__int_as_float(cur) < val) {
        int prev = atomicCAS(ia, cur, __float_as_int(val));
        if (prev == cur) break;
        cur = prev;
    }
}

// pass 1: e = leaky(el[src]+er[dst], 0.2); segment-max into emax[dst]
__global__ void k_edge_logits(const float* __restrict__ el, const float* __restrict__ er,
                              const int* __restrict__ src, const int* __restrict__ dst,
                              float* __restrict__ e, float* __restrict__ emax,
                              int E, int H)
{
    int i = blockIdx.x * blockDim.x + threadIdx.x;
    if (i >= E * H) return;
    int h = i % H, ed = i / H;
    int s = src[ed], d = dst[ed];
    float v = el[s * H + h] + er[d * H + h];
    v = v > 0.f ? v : 0.2f * v;
    e[i] = v;
    atomicMaxF(&emax[d * H + h], v);
}

// pass 2: ex = exp(e - emax[dst]); denom[dst] += ex  (e overwritten in place)
__global__ void k_edge_exp(float* __restrict__ e, const float* __restrict__ emax,
                           const int* __restrict__ dst, float* __restrict__ denom,
                           int E, int H)
{
    int i = blockIdx.x * blockDim.x + threadIdx.x;
    if (i >= E * H) return;
    int h = i % H, ed = i / H;
    int d = dst[ed];
    float ex = __expf(e[i] - emax[d * H + h]);
    e[i] = ex;
    atomicAdd(&denom[d * H + h], ex);
}

// pass 3: agg[dst] += f[src] * (ex / denom[dst])
__global__ void k_edge_agg(const float* __restrict__ e, const float* __restrict__ denom,
                           const float* __restrict__ f,
                           const int* __restrict__ src, const int* __restrict__ dst,
                           float* __restrict__ agg, int E, int H, int D)
{
    int i = blockIdx.x * blockDim.x + threadIdx.x;
    if (i >= E * H) return;
    int h = i % H, ed = i / H;
    int s = src[ed], d = dst[ed];
    float alpha = e[i] / denom[d * H + h];
    const float* fs = f   + (size_t)s * H * D + (size_t)h * D;
    float*       ag = agg + (size_t)d * H * D + (size_t)h * D;
    for (int dd = 0; dd < D; ++dd) atomicAdd(&ag[dd], fs[dd] * alpha);
}

// ELU then recompress to bf16 for the next layer's GEMM input
__global__ void k_elu_to_bf16(const float* __restrict__ agg, bf16* __restrict__ hb, int count) {
    int i = blockIdx.x * blockDim.x + threadIdx.x;
    if (i >= count) return;
    float v = agg[i];
    v = v > 0.f ? v : (__expf(v) - 1.f);
    hb[i] = (bf16)v;
}

// ---------------------------------------------------------------------------
// CNN head: per (node, channel) thread.  ci = logits[paths] is [10 x 5] per
// channel; conv(3x1) over L -> [32 x 3], leaky(0.01); conv(3x1) -> scalar,
// leaky(0.01).  Weights cached in LDS.
// ---------------------------------------------------------------------------
__global__ __launch_bounds__(256) void k_cnn(
    const float* __restrict__ logits, const int* __restrict__ paths,
    const float* __restrict__ w1, const float* __restrict__ b1,
    const float* __restrict__ w2, const float* __restrict__ b2,
    float* __restrict__ out, int N)
{
    __shared__ float s[960 + 32 + 96 + 1];
    for (int i = threadIdx.x; i < 1089; i += blockDim.x) {
        float v;
        if (i < 960)        v = w1[i];
        else if (i < 992)   v = b1[i - 960];
        else if (i < 1088)  v = w2[i - 992];
        else                v = b2[0];
        s[i] = v;
    }
    __syncthreads();
    const float* s_w1 = s;
    const float* s_b1 = s + 960;
    const float* s_w2 = s + 992;
    const float  bias2 = s[1088];

    int c = threadIdx.x & 63;
    int n = blockIdx.x * 4 + (threadIdx.x >> 6);
    if (n >= N) return;

    float ci[10][5];
    const int* pr = paths + (size_t)n * 50;
#pragma unroll
    for (int i = 0; i < 10; ++i)
#pragma unroll
        for (int j = 0; j < 5; ++j)
            ci[i][j] = logits[(size_t)pr[i * 5 + j] * 64 + c];

    float acc = 0.f;
#pragma unroll 4
    for (int co = 0; co < 32; ++co) {
#pragma unroll
        for (int l = 0; l < 3; ++l) {
            float t = s_b1[co];
#pragma unroll
            for (int cin = 0; cin < 10; ++cin)
#pragma unroll
                for (int kh = 0; kh < 3; ++kh)
                    t += ci[cin][l + kh] * s_w1[co * 30 + cin * 3 + kh];
            t = t > 0.f ? t : 0.01f * t;
            acc += t * s_w2[co * 3 + l];
        }
    }
    acc += bias2;
    acc = acc > 0.f ? acc : 0.01f * acc;
    out[(size_t)n * 64 + c] = acc;
}

// ---------------------------------------------------------------------------
extern "C" void kernel_launch(void* const* d_in, const int* in_sizes, int n_in,
                              void* d_out, int out_size, void* d_ws, size_t ws_size,
                              hipStream_t stream)
{
    (void)in_sizes; (void)n_in; (void)out_size; (void)ws_size;

    const float* x    = (const float*)d_in[0];
    const int*   src  = (const int*)  d_in[1];
    const int*   dst  = (const int*)  d_in[2];
    const int*   path = (const int*)  d_in[3];
    const float* W0   = (const float*)d_in[4];
    const float* al0  = (const float*)d_in[5];
    const float* ar0  = (const float*)d_in[6];
    const float* W1   = (const float*)d_in[7];
    const float* al1  = (const float*)d_in[8];
    const float* ar1  = (const float*)d_in[9];
    const float* W2   = (const float*)d_in[10];
    const float* al2  = (const float*)d_in[11];
    const float* ar2  = (const float*)d_in[12];
    const float* cw1  = (const float*)d_in[13];
    const float* cb1  = (const float*)d_in[14];
    const float* cw2  = (const float*)d_in[15];
    const float* cb2  = (const float*)d_in[16];
    float* out = (float*)d_out;

    // workspace carve-out (256B aligned regions)
    char* wsb = (char*)d_ws;
    size_t off = 0;
    auto carve = [&](size_t bytes) -> char* {
        char* p = wsb + off;
        off += (bytes + 255) & ~(size_t)255;
        return p;
    };
    bf16*  hbf   = (bf16*) carve((size_t)NN * 512 * 2);  // GEMM A input (bf16)
    bf16*  Wt    = (bf16*) carve((size_t)128 * 512 * 2); // transposed weights (bf16)
    float* f     = (float*)carve((size_t)NN * 128 * 4);  // GEMM output / features
    float* agg   = (float*)carve((size_t)NN * 128 * 4);  // aggregated output
    float* el    = (float*)carve((size_t)NN * 4 * 4);
    float* er    = (float*)carve((size_t)NN * 4 * 4);
    float* emax  = (float*)carve((size_t)NN * 4 * 4);
    float* denom = (float*)carve((size_t)NN * 4 * 4);
    float* ebuf  = (float*)carve((size_t)EE * 4 * 4);

    auto cdiv = [](int a, int b) { return (a + b - 1) / b; };
    const int gemmWaves = cdiv(NN, 16);          // 3125 (exact)
    const int gemmGrid  = cdiv(gemmWaves, 8);    // 8 waves / block

    // ---------------- layer 0: K=512, N_out=128, H=4, D=32 ----------------
    k_f2bf       <<<cdiv(NN * 512, 256), 256, 0, stream>>>(x, hbf, NN * 512);
    k_wtrans_bf16<<<cdiv(512 * 128, 256), 256, 0, stream>>>(W0, Wt, 512, 128);
    k_gemm_bf16_wmma<8><<<gemmGrid, 256, 0, stream>>>(hbf, Wt, f, NN, 512, 128);
    k_node_prep  <<<cdiv(NN * 4, 256), 256, 0, stream>>>(f, al0, ar0, el, er, emax, denom, agg, NN, 4, 32);
    k_edge_logits<<<cdiv(EE * 4, 256), 256, 0, stream>>>(el, er, src, dst, ebuf, emax, EE, 4);
    k_edge_exp   <<<cdiv(EE * 4, 256), 256, 0, stream>>>(ebuf, emax, dst, denom, EE, 4);
    k_edge_agg   <<<cdiv(EE * 4, 256), 256, 0, stream>>>(ebuf, denom, f, src, dst, agg, EE, 4, 32);
    k_elu_to_bf16<<<cdiv(NN * 128, 256), 256, 0, stream>>>(agg, hbf, NN * 128);

    // ---------------- layer 1: K=128, N_out=128, H=4, D=32 ----------------
    k_wtrans_bf16<<<cdiv(128 * 128, 256), 256, 0, stream>>>(W1, Wt, 128, 128);
    k_gemm_bf16_wmma<8><<<gemmGrid, 256, 0, stream>>>(hbf, Wt, f, NN, 128, 128);
    k_node_prep  <<<cdiv(NN * 4, 256), 256, 0, stream>>>(f, al1, ar1, el, er, emax, denom, agg, NN, 4, 32);
    k_edge_logits<<<cdiv(EE * 4, 256), 256, 0, stream>>>(el, er, src, dst, ebuf, emax, EE, 4);
    k_edge_exp   <<<cdiv(EE * 4, 256), 256, 0, stream>>>(ebuf, emax, dst, denom, EE, 4);
    k_edge_agg   <<<cdiv(EE * 4, 256), 256, 0, stream>>>(ebuf, denom, f, src, dst, agg, EE, 4, 32);
    k_elu_to_bf16<<<cdiv(NN * 128, 256), 256, 0, stream>>>(agg, hbf, NN * 128);

    // ---------------- layer 2: K=128, N_out=64, H=1, D=64 (no activation) --
    k_wtrans_bf16<<<cdiv(128 * 64, 256), 256, 0, stream>>>(W2, Wt, 128, 64);
    k_gemm_bf16_wmma<4><<<gemmGrid, 256, 0, stream>>>(hbf, Wt, f, NN, 128, 64);
    k_node_prep  <<<cdiv(NN * 1, 256), 256, 0, stream>>>(f, al2, ar2, el, er, emax, denom, agg, NN, 1, 64);
    k_edge_logits<<<cdiv(EE * 1, 256), 256, 0, stream>>>(el, er, src, dst, ebuf, emax, EE, 1);
    k_edge_exp   <<<cdiv(EE * 1, 256), 256, 0, stream>>>(ebuf, emax, dst, denom, EE, 1);
    k_edge_agg   <<<cdiv(EE * 1, 256), 256, 0, stream>>>(ebuf, denom, f, src, dst, agg, EE, 1, 64);
    // mean over H=1 heads == identity: agg is logits [NN x 64]

    // ---------------- CNN head ----------------
    k_cnn<<<cdiv(NN, 4), 256, 0, stream>>>(agg, path, cw1, cb1, cw2, cb2, out, NN);
}
